// ReservoirSTDPReadout_78889959293268
// MI455X (gfx1250) — matrix operand order repaired and checked by reading, not compiled, not run on hardware
//
#include <hip/hip_runtime.h>
#include <hip/hip_bf16.h>

// ---------------------------------------------------------------------------
// LIF reservoir + LIF readout, persistent cooperative kernel for gfx1250.
//   v_r(t) = a*v_r(t-1) + x_t@W_in.T + s(t-1)@W_res.T   -> spike, reset
//   v_c(t) = a*v_c(t-1) + s(t)@W_clf.T                  -> spike, reset
// Recurrent GEMM via v_wmma_f32_16x16x32_f16 on [S | x | 0] (K=1056) against
// k-major f16 [W_res | W_in | 0]. State lives in VGPRs for all T steps.
// Latency hiding: (a) 128 GEMM blocks -> ~4 waves/SIMD of SMT, (b) chunked
// double-buffered K-loops pinned with sched_group_barrier (VMEM vs WMMA
// groups) so loads for chunk n+1 issue before chunk n's WMMA chain.
// ---------------------------------------------------------------------------

typedef _Float16 v16h __attribute__((ext_vector_type(16)));
typedef _Float16 v8h  __attribute__((ext_vector_type(8)));
typedef float    v8f  __attribute__((ext_vector_type(8)));

#define BATCH   512
#define NCH     16
#define TSTEPS  2000
#define NRES    1024
#define NCLS    10
#define KEXT    1056          // 1024 spikes + 16 x + 16 zero pad
#define NKT     33            // KEXT / 32
#define ALPHA   0.9f
#define THRESH  1.0f

#define NBLK_GEMM 128
#define NBLK_CLF  4
#define NBLK_WR   1
#define NBLK      (NBLK_GEMM + NBLK_CLF + NBLK_WR)   // 133

// scheduling hint groups: VMEM reads first, then WMMA chain
#define SG_VMEM(n) __builtin_amdgcn_sched_group_barrier(0x020, (n), 0)
#define SG_WMMA(n) __builtin_amdgcn_sched_group_barrier(0x008, (n), 0)

// ws layout (bytes)
#define WS_CNT_OFF   0
#define WS_WB_OFF    256
#define WB_ELEMS     ((size_t)KEXT * NRES)               // 1,081,344
#define WS_WC_OFF    (WS_WB_OFF + WB_ELEMS * 2)
#define WC_ELEMS     ((size_t)NRES * 16)                 // padded classifier
#define WS_S0_OFF    (WS_WC_OFF + WC_ELEMS * 2)
#define S_ELEMS      ((size_t)BATCH * KEXT)              // 540,672
#define WS_S1_OFF    (WS_S0_OFF + S_ELEMS * 2)

// ---------------------------------------------------------------------------
// fragment helpers (ISA 7.12.2 layouts, wave32)
__device__ __forceinline__ v16h ldA(const _Float16* p) {
  v8h lo = *(const v8h*)(p);
  v8h hi = *(const v8h*)(p + 16);
  return __builtin_shufflevector(lo, hi, 0,1,2,3,4,5,6,7,8,9,10,11,12,13,14,15);
}
__device__ __forceinline__ v16h ldB(const _Float16* p) {
  v8h lo = *(const v8h*)(p);
  v8h hi = *(const v8h*)(p + 8);
  return __builtin_shufflevector(lo, hi, 0,1,2,3,4,5,6,7,8,9,10,11,12,13,14,15);
}
__device__ __forceinline__ v8f wmma_f16(v16h a, v16h b, v8f c) {
  return __builtin_amdgcn_wmma_f32_16x16x32_f16(false, a, false, b, (short)0, c,
                                                false, false);
}

// grid-wide barrier: monotonic counter + target count (counter zeroed per launch)
__device__ __forceinline__ void grid_barrier(unsigned* cnt, unsigned target) {
  __threadfence();                 // release: spikes visible at device scope
  __syncthreads();
  if (threadIdx.x == 0) {
    atomicAdd(cnt, 1u);
    while (__hip_atomic_load(cnt, __ATOMIC_RELAXED, __HIP_MEMORY_SCOPE_AGENT) < target)
      __builtin_amdgcn_s_sleep(1);
  }
  __syncthreads();
  __threadfence();                 // acquire: invalidate near caches
}

// ---------------------------------------------------------------------------
// Prologue: pack weights to f16 k-major, init spike/x buffers, zero barrier.
__global__ void __launch_bounds__(256) snn_prep(
    const float* __restrict__ x, const float* __restrict__ W_in,
    const float* __restrict__ W_res, const float* __restrict__ W_clf,
    _Float16* __restrict__ Wb, _Float16* __restrict__ Wc,
    _Float16* __restrict__ S0, _Float16* __restrict__ S1,
    unsigned* __restrict__ cnt) {
  size_t gid = (size_t)blockIdx.x * blockDim.x + threadIdx.x;
  if (gid == 0) *cnt = 0u;
  const size_t total = WB_ELEMS + WC_ELEMS + 2 * S_ELEMS;
  const size_t stride = (size_t)gridDim.x * blockDim.x;
  for (size_t idx = gid; idx < total; idx += stride) {
    if (idx < WB_ELEMS) {
      int kr = (int)(idx / NRES), nr = (int)(idx % NRES);
      int l = kr & 31, s = nr & 15, h = l >> 4;
      int k = (kr & ~31) + (s < 8 ? 8 * h + s : 16 + 8 * h + (s - 8));
      int n = (nr & ~15) + (l & 15);
      float v = 0.0f;
      if (k < NRES)            v = W_res[(size_t)n * NRES + k];   // W_res[n][k]
      else if (k < NRES + NCH) v = W_in[(size_t)n * NCH + (k - NRES)];
      Wb[idx] = (_Float16)v;
    } else if (idx < WB_ELEMS + WC_ELEMS) {
      size_t j = idx - WB_ELEMS;
      int kr = (int)(j / 16), s = (int)(j % 16);
      int l = kr & 31, h = l >> 4;
      int k = (kr & ~31) + (s < 8 ? 8 * h + s : 16 + 8 * h + (s - 8));
      int mm = l & 15;
      float v = (mm < NCLS) ? W_clf[(size_t)mm * NRES + k] : 0.0f;
      Wc[j] = (_Float16)v;
    } else if (idx < WB_ELEMS + WC_ELEMS + S_ELEMS) {
      size_t j = idx - WB_ELEMS - WC_ELEMS;
      int b = (int)(j / KEXT), col = (int)(j % KEXT);
      float v = 0.0f;
      if (col >= NRES && col < NRES + NCH)                 // x_0 columns
        v = x[(size_t)b * NCH * TSTEPS + (size_t)(col - NRES) * TSTEPS];
      S0[j] = (_Float16)v;
    } else {
      S1[idx - WB_ELEMS - WC_ELEMS - S_ELEMS] = (_Float16)0.0f;
    }
  }
}

// ---------------------------------------------------------------------------
__global__ void __launch_bounds__(256) snn_persist(
    const float* __restrict__ x, float* __restrict__ out,
    const _Float16* __restrict__ Wb, const _Float16* __restrict__ Wc,
    _Float16* __restrict__ S0, _Float16* __restrict__ S1,
    unsigned* __restrict__ cnt) {
  const int tid  = threadIdx.x;
  const int lane = tid & 31;
  const int wv   = tid >> 5;
  const int blk  = blockIdx.x;
  const int m    = lane & 15;     // fragment row / col within tile
  const int hi   = lane >> 4;

  if (blk < NBLK_GEMM) {
    // ---------------- reservoir GEMM role: wave owns 16 batch x 32 neurons
    const int cg = blk & 31;            // neuron column group (32 wide)
    const int bq = blk >> 5;            // batch quarter
    const int b0 = (bq * 8 + wv) * 16;  // batch row base
    const int n0 = cg * 32;             // neuron col base
    v8f vst[2] = {};                    // membrane potential (persistent)
    v8f racc[2] = {};                   // spike-count accumulator (persistent)
    const size_t aOff = (size_t)(b0 + m) * KEXT + 8 * hi;   // lane A base (elems)

#pragma unroll 1
    for (int i = 0; i < TSTEPS; ++i) {
      const _Float16* Sc = (i & 1) ? S1 : S0;
      _Float16*       Sn = (i & 1) ? S0 : S1;
      v8f cc[2];
#pragma unroll
      for (int j = 0; j < 2; ++j) cc[j] = vst[j] * ALPHA;   // fuse leak into C seed

      const _Float16* ap = Sc + aOff;
      const _Float16* bp = Wb + (size_t)lane * NRES + n0;

      // chunked double-buffered pipeline: 8 chunks x 4 k-tiles + 1 tail k-tile
      v16h Ab[2][4];
      v16h Bb[2][4][2];
#pragma unroll
      for (int u = 0; u < 4; ++u) {                         // prime chunk 0
        Ab[0][u] = ldA(ap + (size_t)u * 32);
#pragma unroll
        for (int j = 0; j < 2; ++j)
          Bb[0][u][j] = ldB(bp + (size_t)u * 32 * NRES + 16 * j);
      }
      SG_VMEM(24);                                          // prime = 24 b128
#pragma unroll
      for (int ch = 0; ch < 8; ++ch) {
        const int cur = ch & 1, nxt = cur ^ 1;
        if (ch < 7) {                                       // prefetch next chunk
#pragma unroll
          for (int u = 0; u < 4; ++u) {
            const size_t kt = (size_t)(4 * (ch + 1) + u);
            Ab[nxt][u] = ldA(ap + kt * 32);
#pragma unroll
            for (int j = 0; j < 2; ++j)
              Bb[nxt][u][j] = ldB(bp + kt * 32 * NRES + 16 * j);
          }
        } else {                                            // prefetch tail kt=32
          Ab[nxt][0] = ldA(ap + (size_t)32 * 32);
#pragma unroll
          for (int j = 0; j < 2; ++j)
            Bb[nxt][0][j] = ldB(bp + (size_t)32 * 32 * NRES + 16 * j);
        }
#pragma unroll
        for (int u = 0; u < 4; ++u)
#pragma unroll
          for (int j = 0; j < 2; ++j)
            cc[j] = wmma_f16(Ab[cur][u], Bb[cur][u][j], cc[j]);
        if (ch < 7) { SG_VMEM(24); SG_WMMA(8); }            // pin interleave
        else        { SG_VMEM(6);  SG_WMMA(8); }
      }
#pragma unroll
      for (int j = 0; j < 2; ++j)                           // tail k-tile (x | pad)
        cc[j] = wmma_f16(Ab[0][0], Bb[0][0][j], cc[j]);
      SG_WMMA(2);

      // threshold / reset / emit spikes (f16) into next buffer
      _Float16* so = Sn + (size_t)(b0 + 8 * hi) * KEXT + (n0 + m);
#pragma unroll
      for (int j = 0; j < 2; ++j) {
#pragma unroll
        for (int r = 0; r < 8; ++r) {
          float V = cc[j][r];
          float s = (V >= THRESH) ? 1.0f : 0.0f;
          vst[j][r] = V - V * s;        // v * (1 - s)
          racc[j][r] += s;
          so[(size_t)r * KEXT + 16 * j] = (_Float16)s;
        }
      }
      grid_barrier(cnt, (unsigned)(i + 1) * NBLK);
    }
    // write reservoir spike accumulator: out[5120 + b*1024 + n]
    float* ro = out + (size_t)BATCH * NCLS;
#pragma unroll
    for (int j = 0; j < 2; ++j)
#pragma unroll
      for (int r = 0; r < 8; ++r)
        ro[(size_t)(b0 + 8 * hi + r) * NRES + (n0 + 16 * j + m)] = racc[j][r];

  } else if (blk < NBLK_GEMM + NBLK_CLF) {
    // ---------------- classifier role: wave owns 16 batch rows x 16 cols
    const int cw = (blk - NBLK_GEMM) * 8 + wv;   // 0..31 batch tile
    const int b0 = cw * 16;
    v8f vc = {};
    v8f acc = {};
    const size_t aOff = (size_t)(b0 + m) * KEXT + 8 * hi;
    const _Float16* wcb = Wc + (size_t)lane * 16;

    // keep first 16 W_clf fragments resident (128 VGPRs); stream the rest
    v16h wcLo[16];
#pragma unroll
    for (int kt = 0; kt < 16; ++kt)
      wcLo[kt] = ldB(wcb + (size_t)kt * (32 * 16));

#pragma unroll 1
    for (int i = 0; i <= TSTEPS; ++i) {          // lag one step; last pass in epilogue
      const _Float16* Sc = (i & 1) ? S1 : S0;    // holds s(i-1); s(-1)==0 -> no-op
      v8f c = vc * ALPHA;
      const _Float16* ap = Sc + aOff;

      // A stream: 8 chunks x 4 k-tiles, double-buffered
      v16h Ab[2][4];
#pragma unroll
      for (int u = 0; u < 4; ++u) Ab[0][u] = ldA(ap + (size_t)u * 32);
      SG_VMEM(8);
#pragma unroll
      for (int ch = 0; ch < 8; ++ch) {
        const int cur = ch & 1;
        if (ch < 7) {
#pragma unroll
          for (int u = 0; u < 4; ++u)
            Ab[cur ^ 1][u] = ldA(ap + (size_t)(4 * (ch + 1) + u) * 32);
        }
#pragma unroll
        for (int u = 0; u < 4; ++u) {
          const int kt = 4 * ch + u;
          v16h b = (kt < 16) ? wcLo[kt] : ldB(wcb + (size_t)kt * (32 * 16));
          c = wmma_f16(Ab[cur][u], b, c);
        }
        if (ch < 4)      { SG_VMEM(8);  SG_WMMA(4); }
        else if (ch < 7) { SG_VMEM(16); SG_WMMA(4); }
        else             { SG_VMEM(8);  SG_WMMA(4); }
      }
#pragma unroll
      for (int r = 0; r < 8; ++r) {
        float V = c[r];
        float s = (V >= THRESH) ? 1.0f : 0.0f;
        vc[r] = V - V * s;
        acc[r] += s;
      }
      if (i < TSTEPS) grid_barrier(cnt, (unsigned)(i + 1) * NBLK);
    }
    if (m < NCLS) {
#pragma unroll
      for (int r = 0; r < 8; ++r)
        out[(size_t)(b0 + 8 * hi + r) * NCLS + m] = acc[r];
    }

  } else {
    // ---------------- x-writer role: stage x_{i+1} (f32 -> f16) into next buf
#pragma unroll 1
    for (int i = 0; i < TSTEPS; ++i) {
      _Float16* Sn = (i & 1) ? S0 : S1;
      if (i + 1 < TSTEPS) {
        for (int idx = tid; idx < BATCH * NCH; idx += 256) {
          int b = idx >> 4, c = idx & 15;
          float v = x[(size_t)b * NCH * TSTEPS + (size_t)c * TSTEPS + (i + 1)];
          Sn[(size_t)b * KEXT + NRES + c] = (_Float16)v;
        }
      }
      grid_barrier(cnt, (unsigned)(i + 1) * NBLK);
    }
  }
}

// ---------------------------------------------------------------------------
extern "C" void kernel_launch(void* const* d_in, const int* in_sizes, int n_in,
                              void* d_out, int out_size, void* d_ws, size_t ws_size,
                              hipStream_t stream) {
  (void)in_sizes; (void)n_in; (void)out_size; (void)ws_size;
  const float* x     = (const float*)d_in[0];
  const float* W_in  = (const float*)d_in[1];
  const float* W_res = (const float*)d_in[2];
  const float* W_clf = (const float*)d_in[3];

  char* ws = (char*)d_ws;
  unsigned* cnt = (unsigned*)(ws + WS_CNT_OFF);
  _Float16* Wb  = (_Float16*)(ws + WS_WB_OFF);
  _Float16* Wc  = (_Float16*)(ws + WS_WC_OFF);
  _Float16* S0  = (_Float16*)(ws + WS_S0_OFF);
  _Float16* S1  = (_Float16*)(ws + WS_S1_OFF);

  snn_prep<<<2048, 256, 0, stream>>>(x, W_in, W_res, W_clf, Wb, Wc, S0, S1, cnt);
  snn_persist<<<NBLK, 256, 0, stream>>>(x, (float*)d_out, Wb, Wc, S0, S1, cnt);
}
